// Transformer_SparseDec_E_78280073937725
// MI455X (gfx1250) — compile-verified, matrix operand order checked by reading, not run on hardware
//
#include <hip/hip_runtime.h>
#include <hip/hip_bf16.h>
#include <math.h>

typedef __attribute__((ext_vector_type(16))) _Float16 v16h;
typedef __attribute__((ext_vector_type(8)))  float    v8f;
typedef __attribute__((ext_vector_type(4)))  float    vf4;
typedef __attribute__((ext_vector_type(4)))  _Float16 h4;

// ---------------- model constants ----------------
#define BB   32
#define SS   2560
#define EE   256
#define HH_  8
#define HD   32
#define BSW  40      // encoder attention window
#define FFD  1024
#define OSN  24
#define EXPN 5120
#define NBLK 64      // SS / BSW
#define EPSV 1e-5f

// ---------------- small helpers ----------------
__device__ inline v8f vzero() {
  v8f z;
#pragma unroll
  for (int i = 0; i < 8; ++i) z[i] = 0.f;
  return z;
}

__device__ inline v8f wmma_f16(v16h a, v16h b, v8f c) {
  // D = A(16x32 f16) * B(32x16 f16) + C(f32)
  return __builtin_amdgcn_wmma_f32_16x16x32_f16(false, a, false, b, (short)0, c,
                                                false, false);
}

// Load a 16x32-f16 A-fragment (or 32x16 B-fragment, same per-lane pattern) from
// row-major LDS. Per CDNA5 ISA 7.12.2: lane holds row (or col) = lane%16;
// element pair p -> k = 2p + (p>=4 ? 8 : 0) + 8*(lane/16).
__device__ inline v16h load_frag(const _Float16* base, int ld, int row, int grp) {
  v16h f;
#pragma unroll
  for (int p = 0; p < 8; ++p) {
    int kk = 2 * p + ((p & 4) ? 8 : 0) + grp * 8;
    f[2 * p]     = base[row * ld + kk];
    f[2 * p + 1] = base[row * ld + kk + 1];
  }
  return f;
}

__device__ inline h4 cvt4(vf4 v) {
  h4 h;
#pragma unroll
  for (int j = 0; j < 4; ++j) h[j] = (_Float16)v[j];
  return h;
}

__device__ inline float gelu_f(float x) {
  return 0.5f * x * (1.f + erff(x * 0.7071067811865475f));
}

// ---------------- generic WMMA GEMM ----------------
// C[M,NC] = act(A[M,K] @ W[NC,K]^T + bias[NC]) (+ Res[M,NC])
// NC is compile-time -> epilogue row strides become immediate store offsets.
// 128 threads = 4 waves in 2x2 -> 64x64 block tile, each wave 32x32 (2x2 frags)
template <int ACT, int NC>
__global__ __launch_bounds__(128) void k_gemm_wmma(
    const float* __restrict__ A, const float* __restrict__ Wt,
    const float* __restrict__ bias, const float* __restrict__ Res,
    float* __restrict__ C, int M, int K) {
  __shared__ _Float16 sA[64][40];   // 80B row stride: keeps 8B alignment
  __shared__ _Float16 sB[64][40];

  const int tid  = threadIdx.x;
  const int lane = tid & 31, wave = tid >> 5;
  const int wm = (wave >> 1) * 32, wn = (wave & 1) * 32;
  const int lrow = lane & 15, grp = lane >> 4;
  const int bm = blockIdx.y * 64, bn = blockIdx.x * 64;

  v8f acc00 = vzero(), acc01 = vzero(), acc10 = vzero(), acc11 = vzero();

#pragma unroll 1
  for (int k0 = 0; k0 < K; k0 += 32) {
    if (k0 + 32 < K) {  // prefetch next K tile (speculative)
      int pr = bm + wave * 16; if (pr >= M) pr = M - 1;
      int pw = bn + wave * 16; if (pw >= NC) pw = NC - 1;
      __builtin_prefetch(&A[pr * K + k0 + 32], 0, 1);
      __builtin_prefetch(&Wt[pw * K + k0 + 32], 0, 1);
    }
    // stage tiles: 64x32 floats = 512 float4; 4 per thread; b128 load + b64 DS store
#pragma unroll
    for (int it = 0; it < 4; ++it) {
      int idx = tid + it * 128;
      int r = idx >> 3, c4 = (idx & 7) * 4;
      int gr = bm + r;
      vf4 v = {0.f, 0.f, 0.f, 0.f};
      if (gr < M) v = *(const vf4*)(A + gr * K + k0 + c4);
      *(h4*)&sA[r][c4] = cvt4(v);
    }
#pragma unroll
    for (int it = 0; it < 4; ++it) {
      int idx = tid + it * 128;
      int r = idx >> 3, c4 = (idx & 7) * 4;
      int gr = bn + r;
      vf4 v = {0.f, 0.f, 0.f, 0.f};
      if (gr < NC) v = *(const vf4*)(Wt + gr * K + k0 + c4);
      *(h4*)&sB[r][c4] = cvt4(v);
    }
    __syncthreads();

    v16h a0 = load_frag(&sA[0][0], 40, wm + lrow, grp);
    v16h a1 = load_frag(&sA[0][0], 40, wm + 16 + lrow, grp);
    v16h b0 = load_frag(&sB[0][0], 40, wn + lrow, grp);
    v16h b1 = load_frag(&sB[0][0], 40, wn + 16 + lrow, grp);
    acc00 = wmma_f16(a0, b0, acc00);
    acc01 = wmma_f16(a0, b1, acc01);
    acc10 = wmma_f16(a1, b0, acc10);
    acc11 = wmma_f16(a1, b1, acc11);
    __syncthreads();
  }

  // epilogue: lane holds col = lane%16, rows v + 8*(lane/16) within a tile
#pragma unroll
  for (int mt = 0; mt < 2; ++mt) {
#pragma unroll
    for (int nt = 0; nt < 2; ++nt) {
      const v8f acc = (mt == 0) ? (nt == 0 ? acc00 : acc01)
                                : (nt == 0 ? acc10 : acc11);
      const int col  = bn + wn + nt * 16 + lrow;
      const int row0 = bm + wm + mt * 16 + 8 * grp;
      if (col >= NC || row0 >= M) continue;
      const float bv = bias ? bias[col] : 0.f;
      float* Cp = C + row0 * NC + col;
      const float* Rp = Res ? Res + row0 * NC + col : nullptr;
      if (row0 + 7 < M) {  // fast path: immediate-offset stores, no guards
#pragma unroll
        for (int v = 0; v < 8; ++v) {
          float val = acc[v] + bv;
          if (ACT == 1) val = gelu_f(val);
          if (Rp) val += Rp[v * NC];
          Cp[v * NC] = val;
        }
      } else {
#pragma unroll
        for (int v = 0; v < 8; ++v) {
          if (row0 + v < M) {
            float val = acc[v] + bv;
            if (ACT == 1) val = gelu_f(val);
            if (Rp) val += Rp[v * NC];
            Cp[v * NC] = val;
          }
        }
      }
    }
  }
}

// ---------------- fused windowed (block-diagonal) attention ----------------
// One wave per (batch, block, head). W = window, WP = W padded to mult. of 16,
// KP = W padded to mult. of 32 (K dim of the P@V WMMA).
template <int W, int WP, int KP>
__global__ __launch_bounds__(32) void k_block_attn(
    const float* __restrict__ Q, const float* __restrict__ Kp,
    const float* __restrict__ V, float* __restrict__ O, int S_, int NB) {
  __shared__ _Float16 sQ[WP][40];
  __shared__ _Float16 sK[WP][40];
  __shared__ float    sS[WP][WP + 1];
  __shared__ _Float16 sP[WP][KP + 8];
  __shared__ _Float16 sVT[HD][KP + 8];  // V transposed: [dim][key]

  const int lane = threadIdx.x;
  const int hh = blockIdx.x % HH_;
  const int nb = (blockIdx.x / HH_) % NB;
  const int b  = blockIdx.x / (HH_ * NB);
  const int lrow = lane & 15, grp = lane >> 4;
  const int rowbase = (b * S_ + nb * W) * EE + hh * HD;

  // stage Q/K tiles: WP x 32 floats, float4 loads + packed b64 DS stores
#pragma unroll
  for (int it = 0; it < WP / 4; ++it) {
    int idx = lane + it * 32;
    int r = idx >> 3, c4 = (idx & 7) * 4;
    vf4 vq = {0.f, 0.f, 0.f, 0.f}, vk = {0.f, 0.f, 0.f, 0.f};
    if (r < W) {
      vq = *(const vf4*)(Q + rowbase + r * EE + c4);
      vk = *(const vf4*)(Kp + rowbase + r * EE + c4);
    }
    *(h4*)&sQ[r][c4] = cvt4(vq);
    *(h4*)&sK[r][c4] = cvt4(vk);
  }
  for (int r = 0; r < KP; ++r)
    sVT[lane][r] = (r < W) ? (_Float16)V[rowbase + r * EE + lane]
                           : (_Float16)0.f;
  {
    _Float16* p = &sP[0][0];
    for (int idx = lane; idx < WP * (KP + 8); idx += 32) p[idx] = (_Float16)0.f;
  }
  __syncthreads();

  const float scale = 0.17677669529663687f;  // 1/sqrt(32)
#pragma unroll
  for (int mt = 0; mt < WP / 16; ++mt) {
    v16h a = load_frag(&sQ[0][0], 40, mt * 16 + lrow, grp);
#pragma unroll
    for (int nt = 0; nt < WP / 16; ++nt) {
      v16h bfr = load_frag(&sK[0][0], 40, nt * 16 + lrow, grp);
      v8f acc = vzero();
      acc = wmma_f16(a, bfr, acc);
#pragma unroll
      for (int v = 0; v < 8; ++v)
        sS[mt * 16 + v + 8 * grp][nt * 16 + lrow] = acc[v] * scale;
    }
  }
  __syncthreads();

  // row softmax (valid cols 0..W-1 only); write probs as f16 into sP
  for (int r = lane; r < W; r += 32) {
    float m = -1e30f;
    for (int c = 0; c < W; ++c) m = fmaxf(m, sS[r][c]);
    float sum = 0.f;
    for (int c = 0; c < W; ++c) {
      float e = expf(sS[r][c] - m);
      sS[r][c] = e;
      sum += e;
    }
    float inv = 1.f / sum;
    for (int c = 0; c < W; ++c) sP[r][c] = (_Float16)(sS[r][c] * inv);
  }
  __syncthreads();

  // O = P @ V
#pragma unroll
  for (int mt = 0; mt < WP / 16; ++mt) {
    v8f acc0 = vzero(), acc1 = vzero();
#pragma unroll
    for (int ks = 0; ks < KP / 32; ++ks) {
      v16h a  = load_frag(&sP[0][ks * 32], KP + 8, mt * 16 + lrow, grp);
      v16h b0 = load_frag(&sVT[0][ks * 32], KP + 8, lrow, grp);
      v16h b1 = load_frag(&sVT[0][ks * 32], KP + 8, 16 + lrow, grp);
      acc0 = wmma_f16(a, b0, acc0);
      acc1 = wmma_f16(a, b1, acc1);
    }
    const int row0 = mt * 16 + 8 * grp;
    if (row0 >= W) continue;
    float* Op = O + rowbase + row0 * EE + lrow;
    if (row0 + 7 < W) {
#pragma unroll
      for (int v = 0; v < 8; ++v) {
        Op[v * EE]      = acc0[v];
        Op[v * EE + 16] = acc1[v];
      }
    } else {
#pragma unroll
      for (int v = 0; v < 8; ++v) {
        if (row0 + v < W) {
          Op[v * EE]      = acc0[v];
          Op[v * EE + 16] = acc1[v];
        }
      }
    }
  }
}

// ---------------- cross attention (24 queries x SS keys) ----------------
__global__ __launch_bounds__(32) void k_cross_scores(
    const float* __restrict__ Q, const float* __restrict__ Kc,
    float* __restrict__ P) {
  __shared__ _Float16 sQ[32][40];
  __shared__ _Float16 sK[32][40];
  const int lane = threadIdx.x;
  const int bh = blockIdx.y;
  const int b = bh / HH_, hh = bh % HH_;
  const int k0 = blockIdx.x * 32;
  const int lrow = lane & 15, grp = lane >> 4;

#pragma unroll
  for (int it = 0; it < 8; ++it) {
    int idx = lane + it * 32;
    int r = idx >> 3, c4 = (idx & 7) * 4;
    vf4 vq = {0.f, 0.f, 0.f, 0.f};
    if (r < OSN) vq = *(const vf4*)(Q + (b * OSN + r) * EE + hh * HD + c4);
    *(h4*)&sQ[r][c4] = cvt4(vq);
    vf4 vk = *(const vf4*)(Kc + (b * SS + k0 + r) * EE + hh * HD + c4);
    *(h4*)&sK[r][c4] = cvt4(vk);
  }
  __syncthreads();
  const float scale = 0.17677669529663687f;
#pragma unroll
  for (int mt = 0; mt < 2; ++mt) {
    v16h a = load_frag(&sQ[0][0], 40, mt * 16 + lrow, grp);
#pragma unroll
    for (int nt = 0; nt < 2; ++nt) {
      v16h bfr = load_frag(&sK[0][0], 40, nt * 16 + lrow, grp);
      v8f acc = vzero();
      acc = wmma_f16(a, bfr, acc);
      const int row0 = mt * 16 + 8 * grp;
      if (row0 >= OSN) continue;
      float* Pp = P + (bh * OSN + row0) * SS + k0 + nt * 16 + lrow;
      if (row0 + 7 < OSN) {
#pragma unroll
        for (int v = 0; v < 8; ++v) Pp[v * SS] = acc[v] * scale;
      } else {
#pragma unroll
        for (int v = 0; v < 8; ++v)
          if (row0 + v < OSN) Pp[v * SS] = acc[v] * scale;
      }
    }
  }
}

__global__ __launch_bounds__(256) void k_softmax_rows(float* __restrict__ P) {
  __shared__ float red[256];
  float* p = P + blockIdx.x * SS;
  float m = -1e30f;
  for (int c = threadIdx.x; c < SS; c += 256) m = fmaxf(m, p[c]);
  red[threadIdx.x] = m;
  __syncthreads();
  for (int s = 128; s > 0; s >>= 1) {
    if (threadIdx.x < s)
      red[threadIdx.x] = fmaxf(red[threadIdx.x], red[threadIdx.x + s]);
    __syncthreads();
  }
  m = red[0];
  __syncthreads();
  float sum = 0.f;
  for (int c = threadIdx.x; c < SS; c += 256) {
    float e = expf(p[c] - m);
    p[c] = e;
    sum += e;
  }
  red[threadIdx.x] = sum;
  __syncthreads();
  for (int s = 128; s > 0; s >>= 1) {
    if (threadIdx.x < s) red[threadIdx.x] += red[threadIdx.x + s];
    __syncthreads();
  }
  float inv = 1.f / red[0];
  for (int c = threadIdx.x; c < SS; c += 256) p[c] *= inv;
}

__global__ __launch_bounds__(32) void k_cross_out(
    const float* __restrict__ P, const float* __restrict__ V,
    float* __restrict__ O) {
  __shared__ _Float16 sP[32][40];
  __shared__ _Float16 sVT[32][40];
  const int lane = threadIdx.x;
  const int bh = blockIdx.x;
  const int b = bh / HH_, hh = bh % HH_;
  const int lrow = lane & 15, grp = lane >> 4;
  v8f acc00 = vzero(), acc01 = vzero(), acc10 = vzero(), acc11 = vzero();

#pragma unroll 1
  for (int k0 = 0; k0 < SS; k0 += 32) {
    // stage P chunk with float4 loads (rows 24..31 zero)
#pragma unroll
    for (int it = 0; it < 8; ++it) {
      int idx = lane + it * 32;          // 0..255 quads
      int r = idx >> 3, c4 = (idx & 7) * 4;
      vf4 v = {0.f, 0.f, 0.f, 0.f};
      if (r < OSN) v = *(const vf4*)(P + (bh * OSN + r) * SS + k0 + c4);
      *(h4*)&sP[r][c4] = cvt4(v);
    }
    for (int r = 0; r < 32; ++r)
      sVT[lane][r] = (_Float16)V[(b * SS + k0 + r) * EE + hh * HD + lane];
    __syncthreads();
    v16h a0 = load_frag(&sP[0][0], 40, lrow, grp);
    v16h a1 = load_frag(&sP[0][0], 40, 16 + lrow, grp);
    v16h b0 = load_frag(&sVT[0][0], 40, lrow, grp);
    v16h b1 = load_frag(&sVT[0][0], 40, 16 + lrow, grp);
    acc00 = wmma_f16(a0, b0, acc00);
    acc01 = wmma_f16(a0, b1, acc01);
    acc10 = wmma_f16(a1, b0, acc10);
    acc11 = wmma_f16(a1, b1, acc11);
    __syncthreads();
  }
#pragma unroll
  for (int mt = 0; mt < 2; ++mt) {
    const v8f a0 = (mt == 0) ? acc00 : acc10;
    const v8f a1 = (mt == 0) ? acc01 : acc11;
    const int row0 = mt * 16 + 8 * grp;
    if (row0 >= OSN) continue;
    float* Op = O + (b * OSN + row0) * EE + hh * HD + lrow;
    if (row0 + 7 < OSN) {
#pragma unroll
      for (int v = 0; v < 8; ++v) {
        Op[v * EE]      = a0[v];
        Op[v * EE + 16] = a1[v];
      }
    } else {
#pragma unroll
      for (int v = 0; v < 8; ++v) {
        if (row0 + v < OSN) {
          Op[v * EE]      = a0[v];
          Op[v * EE + 16] = a1[v];
        }
      }
    }
  }
}

// ---------------- batchnorm over (B,S) per channel (deterministic) ---------
__global__ __launch_bounds__(256) void k_bn_sum(const float* __restrict__ X,
                                                float* __restrict__ partial,
                                                int rows) {
  int e = threadIdx.x;
  float s = 0.f, ss = 0.f;
  for (int r = blockIdx.x; r < rows; r += gridDim.x) {
    float v = X[r * EE + e];
    s += v;
    ss += v * v;
  }
  partial[blockIdx.x * 512 + e]       = s;
  partial[blockIdx.x * 512 + 256 + e] = ss;
}

__global__ __launch_bounds__(256) void k_bn_reduce(
    const float* __restrict__ partial, float* __restrict__ stats, int nblocks) {
  int i = blockIdx.x * 256 + threadIdx.x;
  if (i >= 512) return;
  float acc = 0.f;
  for (int bkt = 0; bkt < nblocks; ++bkt) acc += partial[bkt * 512 + i];
  stats[i] = acc;
}

__global__ __launch_bounds__(256) void k_bn_apply(
    const float* __restrict__ X, const float* __restrict__ stats,
    const float* __restrict__ g, const float* __restrict__ bb,
    float* __restrict__ Y, int total, float invR) {
  int i = blockIdx.x * 256 + threadIdx.x;
  if (i >= total) return;
  int e = i & (EE - 1);
  float m   = stats[e] * invR;
  float var = stats[256 + e] * invR - m * m;
  Y[i] = (X[i] - m) * rsqrtf(var + EPSV) * g[e] + bb[e];
}

// ---------------- small fused pointwise kernels ----------------
__global__ __launch_bounds__(256) void k_input_pe(
    const float* __restrict__ x, const float* __restrict__ inW,
    const float* __restrict__ inb, const float* __restrict__ pe,
    float* __restrict__ y) {
  int i = blockIdx.x * 256 + threadIdx.x;
  if (i >= BB * SS * EE) return;
  int e = i & (EE - 1);
  int bs = i >> 8;
  int s = bs % SS;
  y[i] = x[bs] * inW[e] + inb[e] + pe[s * EE + e];
}

__global__ __launch_bounds__(256) void k_outer_pe(
    const float* __restrict__ c, const float* __restrict__ aiW,
    const float* __restrict__ aib, const float* __restrict__ pe,
    float* __restrict__ y) {
  int i = blockIdx.x * 256 + threadIdx.x;
  if (i >= BB * OSN * EE) return;
  int e = i & (EE - 1);
  int bo = i >> 8;
  int o = bo % OSN;
  y[i] = c[bo] * aiW[e] + aib[e] + pe[o * EE + e];
}

// out[r] = dot(X[r,0:256], w) + bias[0]
__global__ __launch_bounds__(256) void k_rowdot(const float* __restrict__ X,
                                                const float* __restrict__ w,
                                                const float* __restrict__ bias,
                                                float* __restrict__ out) {
  __shared__ float red[256];
  const int r = blockIdx.x;
  red[threadIdx.x] = X[r * EE + threadIdx.x] * w[threadIdx.x];
  __syncthreads();
  for (int s = 128; s > 0; s >>= 1) {
    if (threadIdx.x < s) red[threadIdx.x] += red[threadIdx.x + s];
    __syncthreads();
  }
  if (threadIdx.x == 0) out[r] = red[0] + bias[0];
}

// ---------------- host orchestration ----------------
extern "C" void kernel_launch(void* const* d_in, const int* in_sizes, int n_in,
                              void* d_out, int out_size, void* d_ws,
                              size_t ws_size, hipStream_t stream) {
  (void)in_sizes; (void)n_in; (void)out_size; (void)ws_size;
  auto F = [&](int i) { return (const float*)d_in[i]; };

  // ---- parameter indices (setup_inputs dict insertion order) ----
  const float* x_in = F(0);
  const float* inW = F(1); const float* inb = F(2); const float* pe = F(3);
  auto EB = [](int l) { return 4 + l * 16; };       // encoder layer base
  auto DB = [](int l) { return 36 + l * 26; };      // decoder layer base
  const int iauxoW = 88, iauxob = 89, iexpW = 90, iexpb = 91;
  const int icmpW = 92, icmpb = 93, iaiW = 94, iaib = 95, ioW = 96, iob = 97;

  // ---- workspace layout (floats) ----
  float* ws = (float*)d_ws;
  const size_t ACT = (size_t)BB * SS * EE;          // 20.97M floats
  float* h   = ws;
  float* t   = ws + ACT;
  float* qb  = ws + 2 * ACT;
  float* kb  = ws + 3 * ACT;
  float* vb  = ws + 4 * ACT;
  float* ob  = ws + 5 * ACT;
  float* ffb = ws + 6 * ACT;                        // B*S*FF floats
  size_t off = 6 * ACT + (size_t)BB * SS * FFD;
  float* part  = ws + off; off += (size_t)256 * 512;
  float* stats = ws + off; off += 512;
  float* s1 = ws + off; off += (size_t)BB * SS;     // [B,S]
  float* s2 = ws + off; off += (size_t)BB * EXPN;   // [B,EXP]
  float* s3 = ws + off; off += 128;                 // [B,OS] (padded)
  const size_t DACT = (size_t)BB * OSN * EE;        // 196608
  float* dbuf = ws + off; off += DACT;
  float* dt   = ws + off; off += DACT;
  float* dq   = ws + off; off += DACT;
  float* dk   = ws + off; off += DACT;
  float* dv   = ws + off; off += DACT;
  float* dob  = ws + off; off += DACT;
  float* dff  = ws + off; off += (size_t)BB * OSN * FFD;
  float* cs   = ffb;  // cross-attn scores [B*H,24,S] reuse FF scratch

  const int Menc = BB * SS;   // 81920
  const int Mdec = BB * OSN;  // 768

  auto gemm = [&](const float* A, const float* Wt, const float* bias,
                  const float* Res, float* C, int M, int N, int K, bool gelu) {
    dim3 g((N + 63) / 64, (M + 63) / 64);
    if (N == 256 && !gelu)
      k_gemm_wmma<0, 256><<<g, 128, 0, stream>>>(A, Wt, bias, Res, C, M, K);
    else if (N == 256 && gelu)
      k_gemm_wmma<1, 256><<<g, 128, 0, stream>>>(A, Wt, bias, Res, C, M, K);
    else if (N == 1024 && !gelu)
      k_gemm_wmma<0, 1024><<<g, 128, 0, stream>>>(A, Wt, bias, Res, C, M, K);
    else if (N == 1024 && gelu)
      k_gemm_wmma<1, 1024><<<g, 128, 0, stream>>>(A, Wt, bias, Res, C, M, K);
    else if (N == 5120 && gelu)
      k_gemm_wmma<1, 5120><<<g, 128, 0, stream>>>(A, Wt, bias, Res, C, M, K);
    else if (N == 24 && !gelu)
      k_gemm_wmma<0, 24><<<g, 128, 0, stream>>>(A, Wt, bias, Res, C, M, K);
  };
  auto tbn = [&](const float* X, const float* g_, const float* b_, float* Y,
                 int rows) {
    k_bn_sum<<<256, 256, 0, stream>>>(X, part, rows);
    k_bn_reduce<<<2, 256, 0, stream>>>(part, stats, 256);
    int total = rows * EE;
    k_bn_apply<<<(total + 255) / 256, 256, 0, stream>>>(X, stats, g_, b_, Y,
                                                        total,
                                                        1.f / (float)rows);
  };

  // ---- input projection + positional encoding ----
  k_input_pe<<<(unsigned)((ACT + 255) / 256), 256, 0, stream>>>(x_in, inW, inb,
                                                                pe, h);

  // ---- encoder (2 layers) ----
  for (int l = 0; l < 2; ++l) {
    int p = EB(l);
    tbn(h, F(p + 8), F(p + 9), t, Menc);                       // n1
    gemm(t, F(p + 0), F(p + 1), nullptr, qb, Menc, EE, EE, false);  // Wq
    gemm(t, F(p + 2), F(p + 3), nullptr, kb, Menc, EE, EE, false);  // Wk
    gemm(t, F(p + 4), F(p + 5), nullptr, vb, Menc, EE, EE, false);  // Wv
    k_block_attn<BSW, 48, 64><<<BB * NBLK * HH_, 32, 0, stream>>>(qb, kb, vb,
                                                                  ob, SS, NBLK);
    gemm(ob, F(p + 6), F(p + 7), h, h, Menc, EE, EE, false);   // Wo + residual
    tbn(h, F(p + 10), F(p + 11), t, Menc);                     // n2
    gemm(t, F(p + 12), F(p + 13), nullptr, ffb, Menc, FFD, EE, true);  // W1+gelu
    gemm(ffb, F(p + 14), F(p + 15), h, h, Menc, EE, FFD, false);       // W2+res
  }
  // h now holds encoder memory (mem)

  // ---- decoder input path ----
  k_rowdot<<<Menc, 256, 0, stream>>>(h, F(iauxoW), F(iauxob), s1);
  gemm(s1, F(iexpW), F(iexpb), nullptr, s2, BB, EXPN, SS, true);   // expand+gelu
  gemm(s2, F(icmpW), F(icmpb), nullptr, s3, BB, OSN, EXPN, false); // compress
  k_outer_pe<<<(unsigned)((DACT + 255) / 256), 256, 0, stream>>>(s3, F(iaiW),
                                                                 F(iaib), pe,
                                                                 dbuf);

  // ---- decoder (2 layers) ----
  for (int l = 0; l < 2; ++l) {
    int p = DB(l);
    // self attention (24 tokens, window == full 24)
    tbn(dbuf, F(p + 8), F(p + 9), dt, Mdec);                   // n1
    gemm(dt, F(p + 0), F(p + 1), nullptr, dq, Mdec, EE, EE, false);
    gemm(dt, F(p + 2), F(p + 3), nullptr, dk, Mdec, EE, EE, false);
    gemm(dt, F(p + 4), F(p + 5), nullptr, dv, Mdec, EE, EE, false);
    k_block_attn<OSN, 32, 32><<<BB * 1 * HH_, 32, 0, stream>>>(dq, dk, dv, dob,
                                                               OSN, 1);
    gemm(dob, F(p + 6), F(p + 7), dbuf, dbuf, Mdec, EE, EE, false);
    // cross attention vs encoder memory
    tbn(dbuf, F(p + 10), F(p + 11), dt, Mdec);                 // n2
    gemm(dt, F(p + 16), F(p + 17), nullptr, dq, Mdec, EE, EE, false);  // cWq
    gemm(h, F(p + 18), F(p + 19), nullptr, kb, Menc, EE, EE, false);   // cWk
    gemm(h, F(p + 20), F(p + 21), nullptr, vb, Menc, EE, EE, false);   // cWv
    k_cross_scores<<<dim3(SS / 32, BB * HH_), 32, 0, stream>>>(dq, kb, cs);
    k_softmax_rows<<<BB * HH_ * OSN, 256, 0, stream>>>(cs);
    k_cross_out<<<BB * HH_, 32, 0, stream>>>(cs, vb, dob);
    gemm(dob, F(p + 22), F(p + 23), dbuf, dbuf, Mdec, EE, EE, false);  // cWo
    // feed-forward
    tbn(dbuf, F(p + 24), F(p + 25), dt, Mdec);                 // n3
    gemm(dt, F(p + 12), F(p + 13), nullptr, dff, Mdec, FFD, EE, true);
    gemm(dff, F(p + 14), F(p + 15), dbuf, dbuf, Mdec, EE, FFD, false);
  }

  // ---- output head ----
  k_rowdot<<<Mdec, 256, 0, stream>>>(dbuf, F(ioW), F(iob), (float*)d_out);
}
